// GCN_83992380441110
// MI455X (gfx1250) — matrix operand order repaired and checked by reading, not compiled, not run on hardware
//
#include <hip/hip_runtime.h>
#include <hip/hip_bf16.h>
#include <math.h>

typedef __attribute__((ext_vector_type(16))) _Float16 v16h;
typedef __attribute__((ext_vector_type(8)))  _Float16 v8h;
typedef __attribute__((ext_vector_type(8)))  float    v8f;

#define N_NODES 100000
#define N_EDGES 800000
#define HD      128
#define IN2D    64
#define BSZ     32
#define TLEN    200

__device__ __forceinline__ float sigmf(float x) { return 1.0f / (1.0f + __expf(-x)); }

// ---------------- WMMA fragment helpers (16x16x32 f16, wave32; CDNA5 ISA layouts) ----
// A (16x32 f16): lanes 0-15 row M=lane, elems 0-7 => K=k0+0..7, elems 8-15 => K=k0+16..23
//                lanes 16-31 row M=lane-16, K=k0+8..15 and k0+24..31
template <int LDA>
__device__ __forceinline__ v16h load_a_global(const float* __restrict__ A,
                                              int row0, int k0, int M, int lane) {
  int row = row0 + (lane & 15);
  int kb  = k0 + ((lane >> 4) << 3);
  v16h a;
  if (row < M) {
    const float* p = A + (size_t)row * LDA + kb;      // 32B aligned (kb % 8 == 0)
    float4 q0 = ((const float4*)p)[0];
    float4 q1 = ((const float4*)p)[1];
    float4 q2 = ((const float4*)(p + 16))[0];
    float4 q3 = ((const float4*)(p + 16))[1];
    a[0] = (_Float16)q0.x;  a[1] = (_Float16)q0.y;  a[2]  = (_Float16)q0.z;  a[3]  = (_Float16)q0.w;
    a[4] = (_Float16)q1.x;  a[5] = (_Float16)q1.y;  a[6]  = (_Float16)q1.z;  a[7]  = (_Float16)q1.w;
    a[8] = (_Float16)q2.x;  a[9] = (_Float16)q2.y;  a[10] = (_Float16)q2.z;  a[11] = (_Float16)q2.w;
    a[12] = (_Float16)q3.x; a[13] = (_Float16)q3.y; a[14] = (_Float16)q3.z;  a[15] = (_Float16)q3.w;
  } else {
#pragma unroll
    for (int i = 0; i < 16; ++i) a[i] = (_Float16)0.0f;
  }
  return a;
}

__device__ __forceinline__ v16h load_a_lds(const float* hs, int ld, int row0, int k0, int lane) {
  int row = row0 + (lane & 15);
  int kb  = k0 + ((lane >> 4) << 3);
  const float* p = hs + row * ld + kb;
  v16h a;
#pragma unroll
  for (int i = 0; i < 8; ++i) a[i] = (_Float16)p[i];
#pragma unroll
  for (int i = 0; i < 8; ++i) a[8 + i] = (_Float16)p[16 + i];
  return a;
}

// B (32x16 f16): lanes 0-15 col N=lane, lanes 16-31 col N=lane-16; K per elem/lane half as in A
// Global loader for [Ncols,K]-stored (transposed) weights: elem (k,n) at B[n*ldb+k]
__device__ __forceinline__ v16h load_bT_global(const float* __restrict__ B, int ldb,
                                               int k0, int col0, int lane) {
  int col = col0 + (lane & 15);
  int kb  = k0 + ((lane >> 4) << 3);
  const float* p = B + (size_t)col * ldb + kb;
  v16h b;
#pragma unroll
  for (int i = 0; i < 16; ++i) b[i] = (_Float16)p[i + ((i >> 3) << 3)];
  return b;
}

// LDS B fragments: bh is column-major f16 [128 cols x KP halves], contiguous k per col.
__device__ __forceinline__ v16h load_b_lds(const _Float16* bh, int KP,
                                           int col0, int k0, int lane) {
  int n  = col0 + (lane & 15);
  int kb = k0 + ((lane >> 4) << 3);
  const _Float16* p = bh + n * KP + kb;               // 16B aligned (KP*2 % 16 == 0)
  v8h lo = *(const v8h*)p;
  v8h hi = *(const v8h*)(p + 16);
  return __builtin_shufflevector(lo, hi, 0, 1, 2, 3, 4, 5, 6, 7,
                                 8, 9, 10, 11, 12, 13, 14, 15);
}

// C/D (16x16 f32): VGPR r: lanes 0-15 -> (M=r, N=lane); lanes 16-31 -> (M=r+8, N=lane-16)
__device__ __forceinline__ void store_c_global(float* __restrict__ C, int ldc, int row0,
                                               int col0, int M, const v8f& c, int lane) {
  int col = col0 + (lane & 15);
  int rb  = row0 + ((lane >> 4) << 3);
#pragma unroll
  for (int r = 0; r < 8; ++r) {
    int row = rb + r;
    if (row < M) C[(size_t)row * ldc + col] = c[r];
  }
}

// ---------------- Dense GEMM: C[M,Ncols] = A[M,K] x B ----------------
// Block = 8 waves; wave w: rows [bx*128 + w*16, +16), cols [by*128, +128).
// B-slice (K x 128) staged once per block into LDS as f16 (fragment-ready, padded),
// then each of the 8 tiles per wave consumes it via ds_load_b128 pairs.
template <bool BT, int K>
__global__ __launch_bounds__(256) void gemm_wmma(const float* __restrict__ A,
                                                 const float* __restrict__ B,
                                                 float* __restrict__ C,
                                                 int M, int ldb, int ldc) {
  constexpr int KP = K + 8;                // pad 8 halves (16B) -> bank rotation
  __shared__ _Float16 bh[128 * KP];        // K=128 -> 34 KB
  int tid  = threadIdx.x;
  int wave = tid >> 5, lane = tid & 31;
  int row0 = blockIdx.x * 128 + wave * 16;
  int col0 = blockIdx.y * 128;

  if (BT) {                                // B stored [Ncols,K]: k-fast reads coalesce
    for (int idx = tid; idx < 128 * K; idx += 256) {
      int n = idx / K, k = idx - n * K;
      bh[n * KP + k] = (_Float16)B[(size_t)(col0 + n) * ldb + k];
    }
  } else {                                 // B stored [K,Ncols]: n-fast reads coalesce
    for (int idx = tid; idx < 128 * K; idx += 256) {
      int k = idx >> 7, n = idx & 127;
      bh[n * KP + k] = (_Float16)B[(size_t)k * ldb + col0 + n];
    }
  }
  __syncthreads();

  v8f acc[8] = {};
#pragma unroll
  for (int k0 = 0; k0 < K; k0 += 32) {
    v16h a = load_a_global<K>(A, row0, k0, M, lane);
#pragma unroll
    for (int j = 0; j < 8; ++j) {
      v16h b = load_b_lds(bh, KP, j * 16, k0, lane);
      acc[j] = __builtin_amdgcn_wmma_f32_16x16x32_f16(false, a, false, b, (short)0,
                                                      acc[j], false, false);
    }
  }
#pragma unroll
  for (int j = 0; j < 8; ++j) store_c_global(C, ldc, row0, col0 + j * 16, M, acc[j], lane);
}

// ---------------- GCN support kernels ----------------
__global__ void deg_accum_kernel(const long long* __restrict__ dst, float* __restrict__ deg, int E) {
  int e = blockIdx.x * blockDim.x + threadIdx.x;
  if (e < E) atomicAdd(&deg[(int)dst[e]], 1.0f);
}

__global__ void dinv_kernel(float* __restrict__ d, int n) {
  int i = blockIdx.x * blockDim.x + threadIdx.x;
  if (i < n) d[i] = rsqrtf(d[i] + 1.0f);   // +1 self-loop
}

// one wave per edge; lane carries 4 features (float4)
__global__ __launch_bounds__(256) void edge_scatter_kernel(const long long* __restrict__ src,
                                                           const long long* __restrict__ dst,
                                                           const float* __restrict__ dinv,
                                                           const float* __restrict__ xw,
                                                           float* __restrict__ agg, int E) {
  int gid = blockIdx.x * blockDim.x + threadIdx.x;
  int e = gid >> 5, lane = gid & 31;
  if (e >= E) return;
  int s = (int)src[e], d = (int)dst[e];
  float nrm = dinv[s] * dinv[d];
  float4 v = ((const float4*)(xw + (size_t)s * HD))[lane];
  float* out = agg + (size_t)d * HD + lane * 4;
  atomicAdd(out + 0, v.x * nrm);
  atomicAdd(out + 1, v.y * nrm);
  atomicAdd(out + 2, v.z * nrm);
  atomicAdd(out + 3, v.w * nrm);
}

__global__ void gcn_finalize_kernel(float* __restrict__ agg, const float* __restrict__ xw,
                                    const float* __restrict__ dinv, const float* __restrict__ bias,
                                    long long total) {
  long long i = (long long)blockIdx.x * blockDim.x + threadIdx.x;
  if (i >= total) return;
  int node = (int)(i >> 7), col = (int)(i & 127);
  float di = dinv[node];
  float v = agg[i] + xw[i] * di * di + bias[col];
  agg[i] = fmaxf(v, 0.0f);
}

// ---------------- LSTM1 recurrence: single persistent block, WMMA per step --------
// 16 waves: wave = (mt, jt); owns batch rows [mt*16,+16) x h-cols [jt*16,+16).
// Computes all 4 gate tiles for its patch (register-resident), so c/h update is local.
// G1[(b*T+t)*512 + col] = x_t @ Wih.T (precomputed) is the initial accumulator.
// (Compiler hoists the 16 Whh fragments into persistent VGPRs across the t-loop.)
__global__ __launch_bounds__(512) void lstm1_kernel(const float* __restrict__ G1,
                                                    const float* __restrict__ Whh,
                                                    const float* __restrict__ bih,
                                                    const float* __restrict__ bhh,
                                                    float* __restrict__ h_out) {
  __shared__ float hs[BSZ * HD];           // 16 KB
  int tid = threadIdx.x, lane = tid & 31, wave = tid >> 5;
  int mt = wave >> 3, jt = wave & 7;
  int row0 = mt * 16, jc0 = jt * 16;
  for (int i = tid; i < BSZ * HD; i += 512) hs[i] = 0.0f;
  int hcol = jc0 + (lane & 15);
  float bI = bih[0 * HD + hcol] + bhh[0 * HD + hcol];
  float bF = bih[1 * HD + hcol] + bhh[1 * HD + hcol];
  float bG = bih[2 * HD + hcol] + bhh[2 * HD + hcol];
  float bO = bih[3 * HD + hcol] + bhh[3 * HD + hcol];
  v8f c = {};
  int rb = row0 + ((lane >> 4) << 3);      // batch-row base in C/D layout
  __syncthreads();
  for (int t = 0; t < TLEN; ++t) {
    v8f g[4];
#pragma unroll
    for (int gi = 0; gi < 4; ++gi) {
      int col = gi * HD + hcol;
#pragma unroll
      for (int r = 0; r < 8; ++r)
        g[gi][r] = G1[((size_t)(rb + r) * TLEN + t) * (4 * HD) + col];
    }
#pragma unroll
    for (int k0 = 0; k0 < HD; k0 += 32) {
      v16h a = load_a_lds(hs, HD, row0, k0, lane);
#pragma unroll
      for (int gi = 0; gi < 4; ++gi) {
        v16h bf = load_bT_global(Whh, HD, k0, gi * HD + jc0, lane);
        g[gi] = __builtin_amdgcn_wmma_f32_16x16x32_f16(false, a, false, bf, (short)0,
                                                       g[gi], false, false);
      }
    }
    __syncthreads();                        // all hs reads complete before overwrite
#pragma unroll
    for (int r = 0; r < 8; ++r) {
      float i_ = sigmf(g[0][r] + bI);
      float f_ = sigmf(g[1][r] + bF);
      float gg = tanhf(g[2][r] + bG);
      float o_ = sigmf(g[3][r] + bO);
      float cc = f_ * c[r] + i_ * gg;
      c[r] = cc;
      hs[(rb + r) * HD + hcol] = o_ * tanhf(cc);
    }
    __syncthreads();
  }
  for (int i = tid; i < BSZ * HD; i += 512) h_out[i] = hs[i];
}

// ---------------- LSTM2: batch=1, 32 steps, tiny -> plain VALU single block -------
__global__ __launch_bounds__(512) void lstm2_kernel(const float* __restrict__ h1,
                                                    const float* __restrict__ Wih,
                                                    const float* __restrict__ Whh,
                                                    const float* __restrict__ bih,
                                                    const float* __restrict__ bhh,
                                                    float* __restrict__ h2_out) {
  __shared__ float h[HD], c[HD], g[4 * HD];
  int tid = threadIdx.x;
  if (tid < HD) { h[tid] = 0.0f; c[tid] = 0.0f; }
  __syncthreads();
  for (int t = 0; t < BSZ; ++t) {
    const float* x = h1 + (size_t)t * HD;
    float acc = bih[tid] + bhh[tid];
    for (int k = 0; k < HD; ++k)
      acc += x[k] * Wih[(size_t)tid * HD + k] + h[k] * Whh[(size_t)tid * HD + k];
    g[tid] = acc;
    __syncthreads();
    if (tid < HD) {
      float i_ = sigmf(g[tid]);
      float f_ = sigmf(g[HD + tid]);
      float gg = tanhf(g[2 * HD + tid]);
      float o_ = sigmf(g[3 * HD + tid]);
      float cc = f_ * c[tid] + i_ * gg;
      c[tid] = cc;
      h[tid] = o_ * tanhf(cc);
    }
    __syncthreads();
  }
  if (tid < HD) h2_out[tid] = h[tid];
}

// ---------------- assemble d = [h2nd[TARGET].flat (1024) , h2 (128)] --------------
__global__ void gather_d_kernel(const float* __restrict__ h, const float* __restrict__ h2,
                                float* __restrict__ d) {
  int i = blockIdx.x * blockDim.x + threadIdx.x;
  if (i < 1024) {
    int node = (i >> 7) * 1000;             // TARGET = 0,1000,...,7000
    d[i] = h[(size_t)node * HD + (i & 127)];
  } else if (i < 1152) {
    d[i] = h2[i - 1024];
  }
}

// ---------------- MLP head: (d@m1_w+m1_b)@m2_w+m2_b, single block -----------------
__global__ __launch_bounds__(576) void mlp_kernel(const float* __restrict__ d,
                                                  const float* __restrict__ m1w,
                                                  const float* __restrict__ m1b,
                                                  const float* __restrict__ m2w,
                                                  const float* __restrict__ m2b,
                                                  float* __restrict__ out) {
  __shared__ float ds[1152];
  __shared__ float red[576];
  int tid = threadIdx.x;
  for (int i = tid; i < 1152; i += 576) ds[i] = d[i];
  __syncthreads();
  float acc = m1b[tid];
  for (int k = 0; k < 1152; ++k) acc += ds[k] * m1w[(size_t)k * 576 + tid];
  red[tid] = acc * m2w[tid];                // m2_w is [576,1]
  __syncthreads();
  if (tid < 64) red[tid] += red[tid + 512];
  __syncthreads();
  for (int s = 256; s > 0; s >>= 1) {
    if (tid < s) red[tid] += red[tid + s];
    __syncthreads();
  }
  if (tid == 0) out[0] = red[0] + m2b[0];
}

extern "C" void kernel_launch(void* const* d_in, const int* in_sizes, int n_in,
                              void* d_out, int out_size, void* d_ws, size_t ws_size,
                              hipStream_t stream) {
  (void)in_sizes; (void)n_in; (void)out_size; (void)ws_size;
  const float* data0 = (const float*)d_in[0];          // [32,200,64]
  const float* data1 = (const float*)d_in[1];          // [100000,128]
  const long long* eidx = (const long long*)d_in[2];   // [2,800000]
  const long long* src = eidx;
  const long long* dst = eidx + N_EDGES;
  const float* W1   = (const float*)d_in[3];
  const float* b1   = (const float*)d_in[4];
  const float* W2   = (const float*)d_in[5];
  const float* b2   = (const float*)d_in[6];
  const float* Wih1 = (const float*)d_in[7];           // [512,64]
  const float* Whh1 = (const float*)d_in[8];           // [512,128]
  const float* bih1 = (const float*)d_in[9];
  const float* bhh1 = (const float*)d_in[10];
  const float* Wih2 = (const float*)d_in[11];          // [512,128]
  const float* Whh2 = (const float*)d_in[12];
  const float* bih2 = (const float*)d_in[13];
  const float* bhh2 = (const float*)d_in[14];
  const float* m1w  = (const float*)d_in[15];          // [1152,576]
  const float* m1b  = (const float*)d_in[16];
  const float* m2w  = (const float*)d_in[17];          // [576,1]
  const float* m2b  = (const float*)d_in[18];

  float* ws = (float*)d_ws;
  size_t off = 0;
  float* dinv = ws + off; off += ((size_t)N_NODES + 63) & ~(size_t)63;
  float* xw   = ws + off; off += (size_t)N_NODES * HD;
  float* aggA = ws + off; off += (size_t)N_NODES * HD;
  float* aggB = ws + off; off += (size_t)N_NODES * HD;
  float* G1   = ws + off; off += (size_t)BSZ * TLEN * 4 * HD;
  float* h1f  = ws + off; off += (size_t)BSZ * HD;
  float* h2f  = ws + off; off += HD;
  float* dvec = ws + off; off += 1152;

  // degree -> dinv (deg starts at 0, +1 self-loop folded into rsqrt)
  hipMemsetAsync(dinv, 0, (size_t)N_NODES * sizeof(float), stream);
  deg_accum_kernel<<<(N_EDGES + 255) / 256, 256, 0, stream>>>(dst, dinv, N_EDGES);
  dinv_kernel<<<(N_NODES + 255) / 256, 256, 0, stream>>>(dinv, N_NODES);

  const long long tot = (long long)N_NODES * HD;
  const int scat_blocks = (int)(((long long)N_EDGES * 32 + 255) / 256);

  // GCN layer 1
  gemm_wmma<false, HD><<<dim3((N_NODES + 127) / 128, 1), 256, 0, stream>>>(data1, W1, xw,
                                                                           N_NODES, HD, HD);
  hipMemsetAsync(aggA, 0, (size_t)tot * sizeof(float), stream);
  edge_scatter_kernel<<<scat_blocks, 256, 0, stream>>>(src, dst, dinv, xw, aggA, N_EDGES);
  gcn_finalize_kernel<<<(int)((tot + 255) / 256), 256, 0, stream>>>(aggA, xw, dinv, b1, tot);

  // GCN layer 2
  gemm_wmma<false, HD><<<dim3((N_NODES + 127) / 128, 1), 256, 0, stream>>>(aggA, W2, xw,
                                                                           N_NODES, HD, HD);
  hipMemsetAsync(aggB, 0, (size_t)tot * sizeof(float), stream);
  edge_scatter_kernel<<<scat_blocks, 256, 0, stream>>>(src, dst, dinv, xw, aggB, N_EDGES);
  gcn_finalize_kernel<<<(int)((tot + 255) / 256), 256, 0, stream>>>(aggB, xw, dinv, b2, tot);

  // LSTM1 input projection: G1[b*T+t, 0:512] = data0[b,t,:] @ Wih1.T  (B transposed)
  gemm_wmma<true, IN2D><<<dim3((BSZ * TLEN + 127) / 128, 4), 256, 0, stream>>>(data0, Wih1, G1,
                                                                               BSZ * TLEN,
                                                                               IN2D, 4 * HD);
  lstm1_kernel<<<1, 512, 0, stream>>>(G1, Whh1, bih1, bhh1, h1f);
  lstm2_kernel<<<1, 512, 0, stream>>>(h1f, Wih2, Whh2, bih2, bhh2, h2f);
  gather_d_kernel<<<5, 256, 0, stream>>>(aggB, h2f, dvec);
  mlp_kernel<<<1, 576, 0, stream>>>(dvec, m1w, m1b, m2w, m2b, (float*)d_out);
}
